// spec_ln_msg_63556926046326
// MI455X (gfx1250) — matrix operand order repaired and checked by reading, not compiled, not run on hardware
//
#include <hip/hip_runtime.h>
#include <math.h>

#define NN 20000
#define KK 512
#define EE 120000
#define ESS 8192

typedef __attribute__((ext_vector_type(16))) __bf16 v16bf;
typedef __attribute__((ext_vector_type(8)))  float  v8f;

__device__ __forceinline__ float lrelu_f(float x, float s) { return x >= 0.f ? x : s * x; }

// ---------------- WMMA fragment loaders (gfx1250 CDNA5 layouts) ----------------
// A 16x32 bf16: lane l -> half=l>>4, row m=l&15.
//   element e of v16: K = (e<8) ? 8*half+e : 16+8*half+(e-8)
__device__ __forceinline__ v16bf load_a_f32(const float* A, long lda, int row, int kb, int half) {
  const float* p0 = A + (long)row * lda + kb + 8 * half;
  const float* p1 = p0 + 16;
  v16bf a;
#pragma unroll
  for (int e = 0; e < 8; ++e) a[e] = (__bf16)p0[e];
#pragma unroll
  for (int e = 0; e < 8; ++e) a[8 + e] = (__bf16)p1[e];
  return a;
}
__device__ __forceinline__ v16bf load_a_bf16(const __bf16* A, long lda, int row, int kb, int half) {
  const __bf16* p0 = A + (long)row * lda + kb + 8 * half;
  const __bf16* p1 = p0 + 16;
  v16bf a;
#pragma unroll
  for (int e = 0; e < 8; ++e) a[e] = p0[e];
#pragma unroll
  for (int e = 0; e < 8; ++e) a[8 + e] = p1[e];
  return a;
}
// B 32x16 bf16: lane l -> col = l&15, half=l>>4; element e: K = (e<8) ? 8*half+e : 16+8*half+(e-8).
// Source stored "W-layout" row-major [Nout, Kdim] (so B[k][n] = src[n][k]): contiguous along K.
__device__ __forceinline__ v16bf load_b_f32(const float* B, long ldb, int col, int kb, int half) {
  const float* p = B + (long)col * ldb + kb + 16 * half;
  v16bf b;
#pragma unroll
  for (int e = 0; e < 16; ++e) b[e] = (__bf16)p[(e < 8 ? e : e + 8) - 8 * half];
  return b;
}
__device__ __forceinline__ v16bf load_b_bf16(const __bf16* B, long ldb, int col, int kb, int half) {
  const __bf16* p = B + (long)col * ldb + kb + 16 * half;
  v16bf b;
#pragma unroll
  for (int e = 0; e < 16; ++e) b[e] = p[(e < 8 ? e : e + 8) - 8 * half];
  return b;
}

// C[m][n] = sum_k A[m][k] * Bsrc[n][k]  (+ bias[n]); C written at column offset coff.
// One wave computes a 16 x (16*NT) output block: one A fragment amortized over NT WMMAs.
// Dtypes are template params so the K-loop is branch-free. All dims multiples of 16/32.
template <bool ABF, bool BBF, int NT>
__global__ void wmma_gemm(const void* A, long lda,
                          const void* B, long ldb,
                          const float* bias,
                          float* C, long ldc, int coff, int Kdim) {
  const int lane = threadIdx.x & 31;
  const int half = lane >> 4;
  const int idx  = lane & 15;
  const int rowA = blockIdx.x * 16 + idx;
  const int nb   = blockIdx.y * NT;
  v8f c[NT] = {};
  for (int kb = 0; kb < Kdim; kb += 32) {
    v16bf a;
    if constexpr (ABF) a = load_a_bf16((const __bf16*)A, lda, rowA, kb, half);
    else               a = load_a_f32((const float*)A, lda, rowA, kb, half);
#pragma unroll
    for (int j = 0; j < NT; ++j) {
      const int colB = (nb + j) * 16 + idx;
      v16bf b;
      if constexpr (BBF) b = load_b_bf16((const __bf16*)B, ldb, colB, kb, half);
      else               b = load_b_f32((const float*)B, ldb, colB, kb, half);
      c[j] = __builtin_amdgcn_wmma_f32_16x16x32_bf16(false, a, false, b, (short)0, c[j],
                                                     false, false);
    }
  }
  const long cb = (long)blockIdx.x * 16;
#pragma unroll
  for (int j = 0; j < NT; ++j) {
    const int colB = (nb + j) * 16 + idx;
    const float bv = bias ? bias[colB] : 0.f;
#pragma unroll
    for (int r = 0; r < 8; ++r)
      C[(cb + r + 8 * half) * ldc + coff + colB] = c[j][r] + bv;
  }
}

// ---------------- scalar helpers ----------------
__device__ float ln64(float v, int t, const float* g, const float* bb, float* red) {
  __syncthreads();
  red[t] = v;
  __syncthreads();
  float s = 0.f;
#pragma unroll 8
  for (int i = 0; i < 64; ++i) s += red[i];
  const float m = s * (1.f / 64.f);
  const float d = v - m;
  __syncthreads();
  red[t] = d * d;
  __syncthreads();
  float s2 = 0.f;
#pragma unroll 8
  for (int i = 0; i < 64; ++i) s2 += red[i];
  return d * rsqrtf(s2 * (1.f / 64.f) + 1e-5f) * g[t] + bb[t];
}

__global__ void encoder_kernel(const float* in, int cin,
                               const float* w1, const float* b1,
                               const float* w2, const float* b2,
                               const float* g, const float* bb,
                               float* out, long ldo, int coff) {
  __shared__ float sin_s[4];
  __shared__ float sz[64];
  __shared__ float red[64];
  const int row = blockIdx.x, t = threadIdx.x;
  if (t < cin) sin_s[t] = in[(long)row * cin + t];
  __syncthreads();
  float h = b1[t];
  for (int i = 0; i < cin; ++i) h += w1[t * cin + i] * sin_s[i];
  h = lrelu_f(h, 0.01f);
  h = ln64(h, t, g, bb, red);
  __syncthreads();
  sz[t] = h;
  __syncthreads();
  float h2 = b2[t];
#pragma unroll 8
  for (int i = 0; i < 64; ++i) h2 += w2[t * 64 + i] * sz[i];
  h2 = lrelu_f(h2, 0.01f);
  h2 = ln64(h2, t, g, bb, red);
  out[(long)row * ldo + coff + t] = h2;
}

__global__ void fill_kernel(float* p, float v, long n) {
  long i = (long)blockIdx.x * blockDim.x + threadIdx.x;
  if (i < n) p[i] = v;
}

// out[c][r] (bf16, [Cw, R]) = in[r*ld_in + c0 + c]
__global__ void tr_cvt_kernel(const float* in, long ld_in, int c0, long R, int Cw, __bf16* out) {
  long i = (long)blockIdx.x * blockDim.x + threadIdx.x;
  if (i >= R * (long)Cw) return;
  long r = i % R;
  long c = i / R;
  out[c * R + r] = (__bf16)in[r * ld_in + c0 + c];
}

__device__ __forceinline__ void atomic_max_f32(float* addr, float v) {
  if (v >= 0.f) atomicMax((int*)addr, __float_as_int(v));
  else          atomicMin((unsigned int*)addr, __float_as_uint(v));
}

__global__ void gat_edge1_kernel(const float* y, const int* src, const int* dst, long EH,
                                 const float* al, const float* ar,
                                 float* aedge, float* amax) {
  long i = (long)blockIdx.x * blockDim.x + threadIdx.x;
  if (i >= EH) return;
  long e = i >> 3;
  int h = (int)(i & 7);
  int s = src[e], d = dst[e];
  const float* ys = y + (long)s * 64 + h * 8;
  const float* yd = y + (long)d * 64 + h * 8;
  float a = 0.f;
#pragma unroll
  for (int c = 0; c < 8; ++c) a += (ys[c] - yd[c]) * al[h * 8 + c] + yd[c] * ar[h * 8 + c];
  a = a >= 0.f ? a : 0.2f * a;
  aedge[i] = a;
  atomic_max_f32(&amax[(long)d * 8 + h], a);
}

__global__ void gat_edge2_kernel(const float* y, const int* src, const int* dst, long EH,
                                 const float* aedge, const float* amax,
                                 float* den, float* acc) {
  long i = (long)blockIdx.x * blockDim.x + threadIdx.x;
  if (i >= EH) return;
  long e = i >> 3;
  int h = (int)(i & 7);
  int s = src[e], d = dst[e];
  const float* ys = y + (long)s * 64 + h * 8;
  const float* yd = y + (long)d * 64 + h * 8;
  float ex = __expf(aedge[i] - amax[(long)d * 8 + h]);
  atomicAdd(&den[(long)d * 8 + h], ex);
#pragma unroll
  for (int c = 0; c < 8; ++c)
    atomicAdd(&acc[(long)d * 64 + h * 8 + c], (ys[c] - yd[c]) * ex);
}

__global__ void gat_final_kernel(const float* acc, const float* den, const float* bvec,
                                 const float* g, const float* bb,
                                 float* out, long ldo, int coff) {
  __shared__ float red[64];
  const int row = blockIdx.x, t = threadIdx.x, h = t >> 3;
  float v = acc[(long)row * 64 + t] / (den[(long)row * 8 + h] + 1e-16f) + bvec[t];
  v = lrelu_f(v, 0.01f);
  v = ln64(v, t, g, bb, red);
  out[(long)row * ldo + coff + t] = v;
}

__global__ void lrelu_ln_kernel(const float* in, long ldi, const float* g, const float* bb,
                                float* out, long ldo) {
  __shared__ float red[64];
  const int row = blockIdx.x, t = threadIdx.x;
  float v = in[(long)row * ldi + t];
  v = lrelu_f(v, 0.01f);
  v = ln64(v, t, g, bb, red);
  out[(long)row * ldo + t] = v;
}

// ---------------- host orchestration ----------------
extern "C" void kernel_launch(void* const* d_in, const int* in_sizes, int n_in,
                              void* d_out, int out_size, void* d_ws, size_t ws_size,
                              hipStream_t stream) {
  (void)in_sizes; (void)n_in; (void)out_size; (void)ws_size;
  const float* x   = (const float*)d_in[0];
  const float* u   = (const float*)d_in[1];
  const float* ev  = (const float*)d_in[2];
  const int*   src = (const int*)d_in[3];
  const int*   dst = (const int*)d_in[4];
  const int*   sps = (const int*)d_in[5];
  const int*   spd = (const int*)d_in[6];
  const float* se1_w = (const float*)d_in[7],  *se1_b = (const float*)d_in[8];
  const float* se2_w = (const float*)d_in[9],  *se2_b = (const float*)d_in[10];
  const float* sp1_w = (const float*)d_in[11], *sp1_b = (const float*)d_in[12];
  const float* sp2_w = (const float*)d_in[13], *sp2_b = (const float*)d_in[14];
  const float* g  = (const float*)d_in[15], *bb = (const float*)d_in[16];
  // GAT param blocks (W, al, ar, b): c1=17, c5=21, c2=25, c6=29, c3=33, c7=37, c8=41
  const float* d1_w = (const float*)d_in[45], *d1_b = (const float*)d_in[46];
  const float* d2_w = (const float*)d_in[47], *d2_b = (const float*)d_in[48];

  char* ws = (char*)d_ws;
  size_t off = 0;
  auto alloc = [&](size_t bytes) -> void* {
    void* p = ws + off;
    off = (off + bytes + 255) & ~(size_t)255;
    return p;
  };
  __bf16* uT   = (__bf16*)alloc((size_t)KK * NN * 2);   // u^T, bf16 [512, 20000]
  float*  Xcat = (float*) alloc((size_t)NN * 960 * 4);  // node features, all stages
  float*  Scat = (float*) alloc((size_t)KK * 640 * 4);  // spectral features
  __bf16* xT   = (__bf16*)alloc((size_t)192 * NN * 2);  // transposed x-feature block
  __bf16* sT   = (__bf16*)alloc((size_t)256 * KK * 2);  // transposed s-feature block
  float*  y    = (float*) alloc((size_t)NN * 64 * 4);   // per-node GAT projection / tmp
  float*  aed  = (float*) alloc((size_t)EE * 8 * 4);    // per-edge attention logits
  float*  amax = (float*) alloc((size_t)NN * 8 * 4);
  float*  den  = (float*) alloc((size_t)NN * 8 * 4);
  float*  acc  = (float*) alloc((size_t)NN * 64 * 4);

  auto cdiv = [](long a, long b) { return (int)((a + b - 1) / b); };

  // One-time bf16 transpose of u -> uT.
  tr_cvt_kernel<<<cdiv((long)NN * KK, 256), 256, 0, stream>>>(u, KK, 0, NN, KK, uT);

  // Encoders -> s0 (Scat[:,0:64]), x0 (Xcat[:,0:64])
  encoder_kernel<<<KK, 64, 0, stream>>>(ev, 1, se1_w, se1_b, se2_w, se2_b, g, bb, Scat, 640, 0);
  encoder_kernel<<<NN, 64, 0, stream>>>(x, 3, sp1_w, sp1_b, sp2_w, sp2_b, g, bb, Xcat, 960, 0);

  auto gat = [&](float* feat, long ld, int cin, int n, const int* sA, const int* dA, long Ecnt,
                 int pidx, int outC0) {
    const float* W  = (const float*)d_in[pidx];
    const float* al = (const float*)d_in[pidx + 1];
    const float* ar = (const float*)d_in[pidx + 2];
    const float* bv = (const float*)d_in[pidx + 3];
    // y = feat[:, :cin] @ W^T   (per-node; lin(xj-xi)=y[src]-y[dst])
    wmma_gemm<false, false, 4><<<dim3(n / 16, 1), 32, 0, stream>>>(
        feat, ld, W, cin, nullptr, y, 64, 0, cin);
    fill_kernel<<<cdiv((long)n * 8, 256), 256, 0, stream>>>(amax, -INFINITY, (long)n * 8);
    fill_kernel<<<cdiv((long)n * 8, 256), 256, 0, stream>>>(den, 0.f, (long)n * 8);
    fill_kernel<<<cdiv((long)n * 64, 256), 256, 0, stream>>>(acc, 0.f, (long)n * 64);
    long EH = Ecnt * 8;
    gat_edge1_kernel<<<cdiv(EH, 256), 256, 0, stream>>>(y, sA, dA, EH, al, ar, aed, amax);
    gat_edge2_kernel<<<cdiv(EH, 256), 256, 0, stream>>>(y, sA, dA, EH, aed, amax, den, acc);
    gat_final_kernel<<<n, 64, 0, stream>>>(acc, den, bv, g, bb, feat, ld, outC0);
  };

  // Scat[:, scoff:scoff+w] = u^T @ Xcat[:, xc0:xc0+w]   (w multiple of 64)
  auto mixS = [&](int xc0, int w, int scoff) {
    tr_cvt_kernel<<<cdiv((long)NN * w, 256), 256, 0, stream>>>(Xcat, 960, xc0, NN, w, xT);
    wmma_gemm<true, true, 4><<<dim3(KK / 16, w / 64), 32, 0, stream>>>(
        uT, NN, xT, NN, nullptr, Scat, 640, scoff, NN);
  };
  // Xcat[:, xcoff:xcoff+w] = u @ Scat[:, sc0:sc0+w]   (w multiple of 64)
  auto mixX = [&](int sc0, int w, int xcoff) {
    tr_cvt_kernel<<<cdiv((long)KK * w, 256), 256, 0, stream>>>(Scat, 640, sc0, KK, w, sT);
    wmma_gemm<false, true, 4><<<dim3(NN / 16, w / 64), 32, 0, stream>>>(
        u, KK, sT, KK, nullptr, Xcat, 960, xcoff, KK);
  };

  // Stage 1: c1/c5 on 64-wide inputs, then mixes (s1, x1 widths 128)
  gat(Scat, 640, 64, KK, sps, spd, ESS, 17, 64);
  gat(Xcat, 960, 64, NN, src, dst, EE, 21, 64);
  mixS(0, 64, 128);
  mixX(0, 64, 128);
  // Stage 2: c2/c6 on 192-wide inputs
  gat(Scat, 640, 192, KK, sps, spd, ESS, 25, 192);
  gat(Xcat, 960, 192, NN, src, dst, EE, 29, 192);
  mixS(64, 128, 256);
  mixX(64, 128, 256);
  // Stage 3: c3/c7 on 384-wide inputs
  gat(Scat, 640, 384, KK, sps, spd, ESS, 33, 384);
  gat(Xcat, 960, 384, NN, src, dst, EE, 37, 384);
  mixS(192, 192, 448);
  mixX(192, 192, 448);
  // Stage 4: c8 on 640-wide node features, final mix (x4 width 320)
  gat(Xcat, 960, 640, NN, src, dst, EE, 41, 640);
  mixX(384, 256, 704);

  // Head: d1 (960->64) + lrelu + LN, then d2 (64->32) -> d_out
  wmma_gemm<false, false, 4><<<dim3(NN / 16, 1), 32, 0, stream>>>(
      Xcat, 960, d1_w, 960, d1_b, acc, 64, 0, 960);
  lrelu_ln_kernel<<<NN, 64, 0, stream>>>(acc, 64, g, bb, y, 64);
  wmma_gemm<false, false, 2><<<dim3(NN / 16, 1), 32, 0, stream>>>(
      y, 64, d2_w, 64, d2_b, (float*)d_out, 32, 0, 64);
}
